// LGVQ_73632919322766
// MI455X (gfx1250) — matrix operand order repaired and checked by reading, not compiled, not run on hardware
//
#include <hip/hip_runtime.h>

// ---------------------------------------------------------------------------
// LGVQ motion encoder for MI455X (gfx1250): conv-downsample + transformer x2,
// then VQ nearest-neighbor. All big GEMMs use v_wmma_f32_16x16x32_f16.
// Round 2: GEMM reworked to 128x64 block tile / 32x32 wave tile with
// double-buffered LDS -> 4 WMMA per frag-load group, 1 barrier per K-step.
// ---------------------------------------------------------------------------

typedef _Float16 f16;
typedef __attribute__((ext_vector_type(16))) _Float16 v16h;
typedef __attribute__((ext_vector_type(8)))  _Float16 v8h;
typedef __attribute__((ext_vector_type(8)))  float    v8f;

#define B_   256
#define D_   512
#define H_   4
#define DH_  128
#define DFF_ 1024
#define KCB_ 8192
#define T0_  196
#define T1_  98
#define T2_  49
#define M1_  (B_ * T1_)   // 25088
#define M2_  (B_ * T2_)   // 12544

static __device__ __forceinline__ float gelu_exact(float x) {
  return 0.5f * x * (1.0f + erff(x * 0.70710678118654752440f));
}

static __device__ __forceinline__ v16h join16(v8h lo, v8h hi) {
  return __builtin_shufflevector(lo, hi, 0,1,2,3,4,5,6,7,8,9,10,11,12,13,14,15);
}

// ---------------------------------------------------------------------------
// Generic WMMA GEMM: C[M,N] = act(A[M,K] @ B[K,N] + bias), A/B f16 row-major.
// Block 256 thr = 8 waves; block tile 128x64; wave tile 32x32 (4 accums).
// Double-buffered LDS staging: one __syncthreads per 32-wide K step.
// M % 128 == 0, N % 64 == 0, K % 32 == 0 guaranteed by caller.
// act: 0=none, 1=relu, 2=gelu.  Cf (f32) and/or Ch (f16) may be null.
// ---------------------------------------------------------------------------
__global__ __launch_bounds__(256)
void k_gemm(const f16* __restrict__ A, const f16* __restrict__ Bm,
            const float* __restrict__ bias,
            float* __restrict__ Cf, f16* __restrict__ Ch,
            int M, int N, int K, int act)
{
  __shared__ __attribute__((aligned(32))) f16 As[2][128][32];  // 16 KB
  __shared__ __attribute__((aligned(32))) f16 Bs[2][32][64];   //  8 KB

  const int tid  = threadIdx.x;
  const int lane = tid & 31;
  const int wave = tid >> 5;
  const int wm   = wave >> 1;          // 0..3 : 32-row slice
  const int wn   = wave & 1;           // 0..1 : 32-col slice
  const int bm   = blockIdx.y << 7;
  const int bn   = blockIdx.x << 6;

  v8f c00 = {}, c01 = {}, c10 = {}, c11 = {};   // [rowHalf][colHalf]

  const int ar = tid >> 1, ac = (tid & 1) << 4;  // A staging: 128x32, v16h each
  const int br = tid >> 3, bc = (tid & 7) << 3;  // B staging: 32x64,  v8h each
  const int m  = lane & 15;                      // A-frag row (ISA layout)
  const int kb = (lane >> 4) << 3;               // A-frag K base (ISA layout)

  auto stage = [&](int p, int k0) {
    *(v16h*)&As[p][ar][ac] = *(const v16h*)&A [(size_t)(bm + ar) * K + k0 + ac];
    *(v8h*) &Bs[p][br][bc] = *(const v8h*) &Bm[(size_t)(k0 + br) * N + bn + bc];
  };
  auto compute = [&](int p) {
    // A fragments: lane holds row m, K in [kb,kb+8) and [kb+16,kb+24)
    v16h a0 = join16(*(const v8h*)&As[p][wm * 32 + m][kb],
                     *(const v8h*)&As[p][wm * 32 + m][kb + 16]);
    v16h a1 = join16(*(const v8h*)&As[p][wm * 32 + 16 + m][kb],
                     *(const v8h*)&As[p][wm * 32 + 16 + m][kb + 16]);
    // B fragments: lane holds column N=m, K in [kb,kb+8) and [kb+16,kb+24)
    v16h b0, b1;
#pragma unroll
    for (int s = 0; s < 16; ++s) {
      int kr = kb + (s & 7) + ((s >> 3) << 4);
      b0[s] = Bs[p][kr][wn * 32 + m];
      b1[s] = Bs[p][kr][wn * 32 + 16 + m];
    }
    c00 = __builtin_amdgcn_wmma_f32_16x16x32_f16(false, a0, false, b0,
                                                 (short)0, c00, false, false);
    c01 = __builtin_amdgcn_wmma_f32_16x16x32_f16(false, a0, false, b1,
                                                 (short)0, c01, false, false);
    c10 = __builtin_amdgcn_wmma_f32_16x16x32_f16(false, a1, false, b0,
                                                 (short)0, c10, false, false);
    c11 = __builtin_amdgcn_wmma_f32_16x16x32_f16(false, a1, false, b1,
                                                 (short)0, c11, false, false);
  };

  stage(0, 0);
  __syncthreads();
  int p = 0;
  for (int k0 = 32; k0 < K; k0 += 32) {
    stage(p ^ 1, k0);     // fill other buffer while computing from this one
    compute(p);
    __syncthreads();
    p ^= 1;
  }
  compute(p);

  // C layout per 16x16 tile: lane l, vgpr j -> (row = j + 8*(l>>4), col = l&15)
#pragma unroll
  for (int rh = 0; rh < 2; ++rh) {
#pragma unroll
    for (int chh = 0; chh < 2; ++chh) {
      v8f acc = rh ? (chh ? c11 : c10) : (chh ? c01 : c00);
      int col = bn + wn * 32 + chh * 16 + (lane & 15);
      float bv = bias ? bias[col] : 0.0f;
#pragma unroll
      for (int j = 0; j < 8; ++j) {
        int row = bm + wm * 32 + rh * 16 + ((lane >> 4) << 3) + j;
        float v = acc[j] + bv;
        if (act == 1)      v = fmaxf(v, 0.0f);
        else if (act == 2) v = gelu_exact(v);
        size_t o = (size_t)row * N + col;
        if (Cf) Cf[o] = v;
        if (Ch) Ch[o] = (f16)v;
      }
    }
  }
}

// ---------------------------------------------------------------------------
// VQ argmin over cc[n] - 2 * x.c  (||x||^2 irrelevant for argmin).
// One wave per 16 rows of X; 16x512 A tile fully register-resident
// (16 fragments); codebookT streamed from L2 (8.4 MB << 192 MB L2).
// ---------------------------------------------------------------------------
__global__ __launch_bounds__(32)
void k_vq(const f16* __restrict__ X,        // [M2_, 512]
          const f16* __restrict__ cbT,      // [512, 8192]  (K-major)
          const float* __restrict__ ccv,    // [8192] ||c||^2
          int* __restrict__ idxOut)
{
  const int lane = threadIdx.x;
  const int r0   = blockIdx.x * 16;
  const int m    = lane & 15;
  const int kb   = (lane >> 4) << 3;

  v16h a[16];
  const f16* arow = X + (size_t)(r0 + m) * D_;
#pragma unroll
  for (int kk = 0; kk < 16; ++kk) {
    a[kk] = join16(*(const v8h*)&arow[kk * 32 + kb],
                   *(const v8h*)&arow[kk * 32 + kb + 16]);
  }

  float minv[8];
  int   mini[8];
#pragma unroll
  for (int j = 0; j < 8; ++j) { minv[j] = 3.4e38f; mini[j] = 0; }

  const int ncol = lane & 15;
  for (int nt = 0; nt < KCB_ / 16; ++nt) {
    v8f c = {};
    const f16* bbase = cbT + nt * 16;
#pragma unroll
    for (int kk = 0; kk < 16; ++kk) {
      // B fragment: lane holds column N=ncol, K = kk*32 + [kb,kb+8) and [kb+16,kb+24)
      v16h b;
#pragma unroll
      for (int s = 0; s < 16; ++s)
        b[s] = bbase[(size_t)(kk * 32 + kb + (s & 7) + ((s >> 3) << 4)) * KCB_ + ncol];
      c = __builtin_amdgcn_wmma_f32_16x16x32_f16(false, a[kk], false, b,
                                                 (short)0, c, false, false);
    }
    int n = nt * 16 + ncol;
    float base = ccv[n];
#pragma unroll
    for (int j = 0; j < 8; ++j) {
      float dist = base - 2.0f * c[j];
      if (dist < minv[j]) { minv[j] = dist; mini[j] = n; }
    }
  }

  // reduce across the 16 lanes of each half (same row, different N residues)
#pragma unroll
  for (int off = 1; off < 16; off <<= 1) {
#pragma unroll
    for (int j = 0; j < 8; ++j) {
      float ov = __shfl_xor(minv[j], off, 32);
      int   oi = __shfl_xor(mini[j], off, 32);
      if (ov < minv[j] || (ov == minv[j] && oi < mini[j])) {
        minv[j] = ov; mini[j] = oi;
      }
    }
  }
  if ((lane & 15) == 0) {
    int half = lane >> 4;
#pragma unroll
    for (int j = 0; j < 8; ++j)
      idxOut[r0 + half * 8 + j] = mini[j];
  }
}

// ---------------------------------------------------------------------------
// Causal attention for one (b, h, q): scores -> softmax -> @V.  128 threads.
// ---------------------------------------------------------------------------
__global__ __launch_bounds__(128)
void k_attn(const float* __restrict__ qkv, float* __restrict__ ao32,
            f16* __restrict__ ao16, int T)
{
  const int q  = blockIdx.x;
  const int bh = blockIdx.y;
  const int b  = bh >> 2, h = bh & 3;
  const int tid = threadIdx.x;

  __shared__ float qv[128];
  __shared__ float s[128];
  __shared__ float red[128];

  qv[tid] = qkv[((size_t)(b * T + q) * 1536) + h * DH_ + tid];
  __syncthreads();

  float sc = -3.0e38f;
  if (tid <= q) {                       // tid <= q < T <= 98 < 128
    const float* kp = qkv + ((size_t)(b * T + tid) * 1536) + 512 + h * DH_;
    float acc = 0.0f;
    for (int d = 0; d < DH_; ++d) acc += qv[d] * kp[d];
    sc = acc * 0.08838834764831845f;    // 1/sqrt(128)
  }
  red[tid] = sc; __syncthreads();
  for (int o = 64; o > 0; o >>= 1) {
    if (tid < o) red[tid] = fmaxf(red[tid], red[tid + o]);
    __syncthreads();
  }
  float mx = red[0];
  __syncthreads();
  float e = (tid <= q) ? expf(sc - mx) : 0.0f;
  s[tid] = e; red[tid] = e; __syncthreads();
  for (int o = 64; o > 0; o >>= 1) {
    if (tid < o) red[tid] += red[tid + o];
    __syncthreads();
  }
  float inv = 1.0f / red[0];

  float acc = 0.0f;
  for (int k = 0; k <= q; ++k)
    acc += s[k] * qkv[((size_t)(b * T + k) * 1536) + 1024 + h * DH_ + tid];
  float out = acc * inv;
  size_t oi = ((size_t)(b * T + q) * D_) + h * DH_ + tid;
  ao32[oi] = out;
  ao16[oi] = (f16)out;
}

// ---------------------------------------------------------------------------
// y = LayerNorm(xin + res) * g + b, writes f32 + f16.  In-place safe
// (res/y32 may alias: each element read before write by its own thread).
// ---------------------------------------------------------------------------
__global__ __launch_bounds__(128)
void k_add_ln(const float* xin, const float* res,
              const float* __restrict__ g, const float* __restrict__ bb,
              float* y32, f16* __restrict__ y16)
{
  const int row = blockIdx.x, tid = threadIdx.x;
  __shared__ float red[128];
  const float* xr = xin + (size_t)row * D_;
  const float* rr = res + (size_t)row * D_;
  float v[4]; float s = 0.0f;
#pragma unroll
  for (int i = 0; i < 4; ++i) { v[i] = xr[tid + i * 128] + rr[tid + i * 128]; s += v[i]; }
  red[tid] = s; __syncthreads();
  for (int o = 64; o > 0; o >>= 1) { if (tid < o) red[tid] += red[tid + o]; __syncthreads(); }
  float mean = red[0] * (1.0f / D_);
  __syncthreads();
  float vs = 0.0f;
#pragma unroll
  for (int i = 0; i < 4; ++i) { float d = v[i] - mean; vs += d * d; }
  red[tid] = vs; __syncthreads();
  for (int o = 64; o > 0; o >>= 1) { if (tid < o) red[tid] += red[tid + o]; __syncthreads(); }
  float inv = rsqrtf(red[0] * (1.0f / D_) + 1e-5f);
#pragma unroll
  for (int i = 0; i < 4; ++i) {
    int c = tid + i * 128;
    float o = (v[i] - mean) * inv * g[c] + bb[c];
    size_t oi = (size_t)row * D_ + c;
    y32[oi] = o;
    y16[oi] = (f16)o;
  }
}

// ------------------------- small helper kernels ----------------------------
__global__ void k_cast16(const float* __restrict__ src, f16* __restrict__ dst, int n) {
  int i = blockIdx.x * blockDim.x + threadIdx.x;
  if (i < n) dst[i] = (f16)src[i];
}

// im2col for causal stride-2 conv, repeat-first-frame pad of 2.
__global__ void k_im2col(const float* __restrict__ x, f16* __restrict__ col,
                         int Tin, int Tout)
{
  size_t i = (size_t)blockIdx.x * blockDim.x + threadIdx.x;
  size_t total = (size_t)B_ * Tout * 3 * D_;
  if (i >= total) return;
  int d = (int)(i % D_);
  int h = (int)((i / D_) % 3);
  int t = (int)((i / ((size_t)3 * D_)) % Tout);
  int b = (int)(i / ((size_t)3 * D_ * Tout));
  int st = 2 * t + h - 2; if (st < 0) st = 0;
  col[i] = (f16)x[((size_t)b * Tin + st) * D_ + d];
}

// conv_w[o][i][h] (O,I,3) -> wp[h*D + i][o]  (row-major [3D, D] f16)
__global__ void k_wpack(const float* __restrict__ wsrc, f16* __restrict__ wdst) {
  int i = blockIdx.x * blockDim.x + threadIdx.x;   // i = row*512 + o
  if (i >= D_ * D_ * 3) return;
  int o   = i % D_;
  int row = i / D_;
  int ii  = row % D_;
  int h   = row / D_;
  wdst[i] = (f16)wsrc[((size_t)o * D_ + ii) * 3 + h];
}

// codebook [K,512] f32 -> cbT [512,K] f16
__global__ void k_cbT(const float* __restrict__ cb, f16* __restrict__ cbT) {
  int i = blockIdx.x * blockDim.x + threadIdx.x;   // i = k*8192 + n
  if (i >= D_ * KCB_) return;
  int n = i % KCB_;
  int k = i / KCB_;
  cbT[i] = (f16)cb[(size_t)n * D_ + k];
}

__global__ __launch_bounds__(128)
void k_rowsq(const float* __restrict__ cb, float* __restrict__ out) {
  int n = blockIdx.x, tid = threadIdx.x;
  __shared__ float red[128];
  const float* r = cb + (size_t)n * D_;
  float s = 0.0f;
  for (int i = tid; i < D_; i += 128) { float v = r[i]; s += v * v; }
  red[tid] = s; __syncthreads();
  for (int o = 64; o > 0; o >>= 1) { if (tid < o) red[tid] += red[tid + o]; __syncthreads(); }
  if (tid == 0) out[n] = red[0];
}

__global__ void k_gather(const float* __restrict__ cb, const int* __restrict__ idx,
                         float* __restrict__ q) {
  size_t i = (size_t)blockIdx.x * blockDim.x + threadIdx.x;
  if (i >= (size_t)M2_ * D_) return;
  int row = (int)(i / D_);
  int d   = (int)(i % D_);
  q[i] = cb[(size_t)idx[row] * D_ + d];
}

// ---------------------------------------------------------------------------
// Host orchestration.  Workspace layout (~447 MB).
// ---------------------------------------------------------------------------
extern "C" void kernel_launch(void* const* d_in, const int* in_sizes, int n_in,
                              void* d_out, int out_size, void* d_ws, size_t ws_size,
                              hipStream_t stream)
{
  (void)in_sizes; (void)n_in; (void)out_size; (void)ws_size;

  const float* motion   = (const float*)d_in[0];
  const float* conv_w   = (const float*)d_in[1];
  const float* conv_b   = (const float*)d_in[2];
  const float* wqkv     = (const float*)d_in[3];
  const float* bqkv     = (const float*)d_in[4];
  const float* wo       = (const float*)d_in[5];
  const float* bo       = (const float*)d_in[6];
  const float* ln1_g    = (const float*)d_in[7];
  const float* ln1_b    = (const float*)d_in[8];
  const float* ln2_g    = (const float*)d_in[9];
  const float* ln2_b    = (const float*)d_in[10];
  const float* w1       = (const float*)d_in[11];
  const float* b1       = (const float*)d_in[12];
  const float* w2       = (const float*)d_in[13];
  const float* b2       = (const float*)d_in[14];
  const float* codebook = (const float*)d_in[15];

  char* wp = (char*)d_ws;
  auto alloc = [&](size_t bytes) -> char* {
    char* p = wp; wp += (bytes + 255) & ~(size_t)255; return p;
  };
  float* x32   = (float*)alloc((size_t)M1_ * D_ * 4);
  f16*   x16   = (f16*)  alloc((size_t)M1_ * D_ * 2);
  f16*   col16 = (f16*)  alloc((size_t)M1_ * 3 * D_ * 2);   // also FFN hidden
  float* qkv32 = (float*)alloc((size_t)M1_ * 3 * D_ * 4);
  float* ao32  = (float*)alloc((size_t)M1_ * D_ * 4);
  f16*   ao16  = (f16*)  alloc((size_t)M1_ * D_ * 2);
  float* tmp32 = (float*)alloc((size_t)M1_ * D_ * 4);       // proj / ffn out
  f16*   wbuf  = (f16*)  alloc((size_t)D_ * 3 * D_ * 2);    // staged f16 weights
  f16*   cbT   = (f16*)  alloc((size_t)D_ * KCB_ * 2);
  float* ccv   = (float*)alloc((size_t)KCB_ * 4);

  for (int l = 0; l < 2; ++l) {
    const int Tin  = l ? T1_ : T0_;
    const int Tout = l ? T2_ : T1_;
    const int Mq   = B_ * Tout;
    const float* xin = l ? x32 : motion;

    // conv downsample as GEMM: [Mq,1536] @ [1536,512] + b, GELU
    size_t nIm = (size_t)Mq * 3 * D_;
    k_im2col<<<(unsigned)((nIm + 255) / 256), 256, 0, stream>>>(xin, col16, Tin, Tout);
    k_wpack<<<(D_ * D_ * 3 + 255) / 256, 256, 0, stream>>>(
        conv_w + (size_t)l * D_ * D_ * 3, wbuf);
    k_gemm<<<dim3(D_ / 64, Mq / 128), 256, 0, stream>>>(
        col16, wbuf, conv_b + l * D_, x32, x16, Mq, D_, 3 * D_, /*gelu*/2);

    // QKV: [Mq,512] @ [512,1536] + b
    k_cast16<<<(D_ * 3 * D_ + 255) / 256, 256, 0, stream>>>(
        wqkv + (size_t)l * D_ * 3 * D_, wbuf, D_ * 3 * D_);
    k_gemm<<<dim3(3 * D_ / 64, Mq / 128), 256, 0, stream>>>(
        x16, wbuf, bqkv + l * 3 * D_, qkv32, (f16*)nullptr, Mq, 3 * D_, D_, 0);

    // causal attention
    k_attn<<<dim3(Tout, B_ * H_), 128, 0, stream>>>(qkv32, ao32, ao16, Tout);

    // output proj + residual + LN1
    k_cast16<<<(D_ * D_ + 255) / 256, 256, 0, stream>>>(
        wo + (size_t)l * D_ * D_, wbuf, D_ * D_);
    k_gemm<<<dim3(D_ / 64, Mq / 128), 256, 0, stream>>>(
        ao16, wbuf, bo + l * D_, tmp32, (f16*)nullptr, Mq, D_, D_, 0);
    k_add_ln<<<Mq, 128, 0, stream>>>(tmp32, x32, ln1_g + l * D_, ln1_b + l * D_,
                                     x32, x16);

    // FFN: relu(x@w1+b1)@w2+b2, residual + LN2
    f16* h16 = col16;
    k_cast16<<<(D_ * DFF_ + 255) / 256, 256, 0, stream>>>(
        w1 + (size_t)l * D_ * DFF_, wbuf, D_ * DFF_);
    k_gemm<<<dim3(DFF_ / 64, Mq / 128), 256, 0, stream>>>(
        x16, wbuf, b1 + l * DFF_, (float*)nullptr, h16, Mq, DFF_, D_, /*relu*/1);
    k_cast16<<<(DFF_ * D_ + 255) / 256, 256, 0, stream>>>(
        w2 + (size_t)l * DFF_ * D_, wbuf, DFF_ * D_);
    k_gemm<<<dim3(D_ / 64, Mq / 128), 256, 0, stream>>>(
        h16, wbuf, b2 + l * D_, tmp32, (f16*)nullptr, Mq, D_, DFF_, 0);
    k_add_ln<<<Mq, 128, 0, stream>>>(tmp32, x32, ln2_g + l * D_, ln2_b + l * D_,
                                     x32, x16);
  }

  // VQ: argmin_n (||c_n||^2 - 2 x.c_n), quant = codebook[idx]
  k_cbT<<<(D_ * KCB_ + 255) / 256, 256, 0, stream>>>(codebook, cbT);
  k_rowsq<<<KCB_, 128, 0, stream>>>(codebook, ccv);

  float* qout  = (float*)d_out;                    // [M2_, 512] f32
  int*   idxo  = (int*)(qout + (size_t)M2_ * D_);  // [M2_] int32
  k_vq<<<M2_ / 16, 32, 0, stream>>>(x16, cbT, ccv, idxo);
  k_gather<<<(unsigned)(((size_t)M2_ * D_ + 255) / 256), 256, 0, stream>>>(
      codebook, idxo, qout);
}
